// BidirectionalRNN_23149873725701
// MI455X (gfx1250) — compile-verified, matrix operand order checked by reading, not run on hardware
//
#include <hip/hip_runtime.h>
#include <stdint.h>

#define B_   32
#define S_   512
#define I_   1024
#define H_   1024

typedef __bf16 v16bf __attribute__((ext_vector_type(16)));
typedef float  v8f   __attribute__((ext_vector_type(8)));
typedef unsigned u32x4 __attribute__((ext_vector_type(4)));
typedef int      i32x4 __attribute__((ext_vector_type(4)));
typedef int      i32x8 __attribute__((ext_vector_type(8)));

typedef int v4i __attribute__((ext_vector_type(4)));
typedef __attribute__((address_space(1))) v4i g_v4i;   // global int4
typedef __attribute__((address_space(3))) v4i l_v4i;   // LDS int4

union FragBF { v16bf v; uint4 q[2]; };

#if __has_builtin(__builtin_amdgcn_global_load_async_to_lds_b128)
#define HAVE_ASYNC_LDS 1
#else
#define HAVE_ASYNC_LDS 0
#endif
#if __has_builtin(__builtin_amdgcn_tensor_load_to_lds)
#define HAVE_TDM 1
#else
#define HAVE_TDM 0
#endif

__device__ __forceinline__ unsigned short f2bf_raw(float f) {
  union { float f; unsigned u; } v; v.f = f;
  unsigned r = v.u + 0x7FFFu + ((v.u >> 16) & 1u);   // round-to-nearest-even
  return (unsigned short)(r >> 16);
}

// 16x32 bf16 WMMA fragment from row-major (row, K) storage (A layout; B dual
// layout is identical when B is stored (N,K) row-major, as W_ih/W_hh are).
// CDNA5 ISA: lanes 0-15 rows, VGPR0-3 = K0..7, VGPR4-7 = K16..23;
// lanes 16-31 shift K by +8.  Two contiguous 16B chunks per lane.
__device__ __forceinline__ v16bf load_frag(const unsigned short* base, int ld, int lane) {
  const int half = (lane >> 4) & 1;
  const int r    = lane & 15;
  const unsigned short* p = base + (size_t)r * ld + half * 8;
  FragBF f;
  f.q[0] = *(const uint4*)(p);
  f.q[1] = *(const uint4*)(p + 16);
  return f.v;
}

// ---------------------------------------------------------------- conversions
__global__ void k_f32_to_bf16(const float* __restrict__ src,
                              unsigned short* __restrict__ dst, int n) {
  int i = blockIdx.x * blockDim.x + threadIdx.x;
  if (i < n) dst[i] = f2bf_raw(src[i]);
}

// ------------------------------------------------- phase A: xp = x*W_ih^T + b
__global__ __launch_bounds__(256)
void k_gemm_xp(const unsigned short* __restrict__ xbf,      // (B_*S_, I_)
               const unsigned short* __restrict__ wihf_bf,  // (H_, I_)
               const unsigned short* __restrict__ wihb_bf,
               const float* __restrict__ bih_f, const float* __restrict__ bhh_f,
               const float* __restrict__ bih_b, const float* __restrict__ bhh_b,
               float* __restrict__ xp_f, float* __restrict__ xp_b)
{
  const int dir = blockIdx.z;
  const unsigned short* W = dir ? wihb_bf : wihf_bf;
  const float* bih = dir ? bih_b : bih_f;
  const float* bhh = dir ? bhh_b : bhh_f;
  float* xp = dir ? xp_b : xp_f;

  const int lane = threadIdx.x & 31;
  const int wave = threadIdx.x >> 5;   // 0..7
  const int wm   = wave >> 1;          // 0..3
  const int wn   = wave & 1;           // 0..1

  const int m0 = blockIdx.x * 128 + wm * 32;
  const int n0 = blockIdx.y * 64  + wn * 32;

  v8f acc00 = {}, acc01 = {}, acc10 = {}, acc11 = {};

  const unsigned short* Arow0 = xbf + (size_t)m0 * I_;
  const unsigned short* Arow1 = xbf + (size_t)(m0 + 16) * I_;
  const unsigned short* Brow0 = W   + (size_t)n0 * I_;
  const unsigned short* Brow1 = W   + (size_t)(n0 + 16) * I_;

  for (int k0 = 0; k0 < I_; k0 += 32) {
    if (k0 + 32 < I_) {                       // global_prefetch_b8 next K slab
      __builtin_prefetch(Arow0 + k0 + 32, 0, 1);
      __builtin_prefetch(Brow0 + k0 + 32, 0, 1);
    }
    v16bf a0 = load_frag(Arow0 + k0, I_, lane);
    v16bf a1 = load_frag(Arow1 + k0, I_, lane);
    v16bf b0 = load_frag(Brow0 + k0, I_, lane);
    v16bf b1 = load_frag(Brow1 + k0, I_, lane);
    acc00 = __builtin_amdgcn_wmma_f32_16x16x32_bf16(false, a0, false, b0, (short)0, acc00, false, false);
    acc01 = __builtin_amdgcn_wmma_f32_16x16x32_bf16(false, a0, false, b1, (short)0, acc01, false, false);
    acc10 = __builtin_amdgcn_wmma_f32_16x16x32_bf16(false, a1, false, b0, (short)0, acc10, false, false);
    acc11 = __builtin_amdgcn_wmma_f32_16x16x32_bf16(false, a1, false, b1, (short)0, acc11, false, false);
  }

  const int col  = lane & 15;
  const int rowb = (lane >> 4) * 8;
  const float bias0 = bih[n0 + col]      + bhh[n0 + col];
  const float bias1 = bih[n0 + 16 + col] + bhh[n0 + 16 + col];
#pragma unroll
  for (int r = 0; r < 8; ++r) {
    const int ra = m0 + rowb + r;
    const int rb = m0 + 16 + rowb + r;
    xp[(size_t)ra * H_ + n0 + col]      = acc00[r] + bias0;
    xp[(size_t)ra * H_ + n0 + 16 + col] = acc01[r] + bias1;
    xp[(size_t)rb * H_ + n0 + col]      = acc10[r] + bias0;
    xp[(size_t)rb * H_ + n0 + 16 + col] = acc11[r] + bias1;
  }
}

// -------------------------------------------------- phase B: recurrent scan
// Padded LDS layout matching the TDM pad pattern: pad 4 dwords (8 bf16) after
// every 256 dwords (512 bf16) -> row stride 520 dwords = 1040 bf16 = 2080 B
// (16B aligned); element k sits at k + (k>=512 ? 8 : 0) within its row.
#define LDWP 1040
#define HOFF (32 * LDWP)          // h region follows W region (bf16 elems)

__device__ __forceinline__ int padk(int k) { return k + ((k >= 512) ? 8 : 0); }

__device__ __forceinline__ v16bf load_frag_lds(const unsigned short* lds_base,
                                               int rowbase, int k0, int lane) {
  const int half = (lane >> 4) & 1;
  const int r    = lane & 15;
  const unsigned short* p = lds_base + (rowbase + r) * LDWP + padk(k0) + half * 8;
  FragBF f;
  f.q[0] = *(const uint4*)(p);
  f.q[1] = *(const uint4*)(p + 16);
  return f.v;
}

__global__ __launch_bounds__(128)
void k_scan(const unsigned short* __restrict__ whhf_bf,
            const unsigned short* __restrict__ whhb_bf,
            const float* __restrict__ xp_f, const float* __restrict__ xp_b,
            unsigned short* __restrict__ hbuf,   // [dir][2][B_*H_] bf16
            unsigned* __restrict__ bars,         // 2 counters, zeroed
            float* __restrict__ out)             // (B_, S_, 2*H_)
{
  const int dir  = blockIdx.x >> 5;
  const int blk  = blockIdx.x & 31;
  const int n0   = blk * 32;
  const int lane = threadIdx.x & 31;
  const int wave = threadIdx.x >> 5;   // 0..3
  const int wm   = wave >> 1;          // batch tile
  const int wn   = wave & 1;           // column tile

  const unsigned short* Whh = dir ? whhb_bf : whhf_bf;
  const float* xp = dir ? xp_b : xp_f;
  unsigned short* h0 = hbuf + (size_t)dir * 2 * (B_ * H_);
  unsigned short* h1 = h0 + (B_ * H_);
  unsigned* bar = bars + dir;

  extern __shared__ unsigned short Lds[];        // [0,HOFF): W  [HOFF,2*HOFF): h

  // ---- stage this block's 32 rows of W_hh into LDS via TDM ----
#if HAVE_TDM
  if (wave == 0) {
    const unsigned long long ga = (unsigned long long)(uintptr_t)(Whh + (size_t)n0 * I_);
    const unsigned lds_off = (unsigned)(uintptr_t)&Lds[0];   // flat LDS addr[31:0] = LDS offset
    u32x4 g0;
    g0[0] = 1u;                                  // count=1, no gather
    g0[1] = lds_off;                             // lds_addr
    g0[2] = (unsigned)(ga & 0xFFFFFFFFu);        // global_addr[31:0]
    g0[3] = (unsigned)((ga >> 32) & 0x01FFFFFFu) | (2u << 30);  // global_addr[56:32] | type=2
    i32x8 g1;
    g1[0] = (1 << 16)       // data_size = 2B
          | (1 << 20)       // pad_enable
          | (7 << 22)       // pad_interval: 256 dwords
          | (3 << 25);      // pad_amount: 4 dwords
    g1[1] = (int)((unsigned)(I_ & 0xFFFF) << 16);   // tensor_dim0[15:0] @ bits 63:48
    g1[2] = (int)((unsigned)(H_ & 0xFFFF) << 16);   // tensor_dim1[15:0] @ bits 95:80 (dim0 hi16 = 0)
    g1[3] = (int)((unsigned)I_ << 16);              // tile_dim0 = 1024 @ bits 127:112
    g1[4] = 32;                                     // tile_dim1 = 32 rows
    g1[5] = I_;                                     // tensor_dim0_stride[31:0]
    g1[6] = 0;
    g1[7] = 0;
    i32x4 gz = {0, 0, 0, 0};
#if defined(__clang_major__) && (__clang_major__ >= 23)
    i32x8 gz8 = {0, 0, 0, 0, 0, 0, 0, 0};
    __builtin_amdgcn_tensor_load_to_lds(g0, g1, gz, gz, gz8, 0);
#else
    __builtin_amdgcn_tensor_load_to_lds(g0, g1, gz, gz, 0);
#endif
    __builtin_amdgcn_s_wait_tensorcnt(0);
  }
#else
  for (int idx = threadIdx.x; idx < 32 * (I_ / 8); idx += blockDim.x) {
    const int rr = idx >> 7;
    const int k  = (idx & 127) * 8;
    *(uint4*)&Lds[rr * LDWP + padk(k)] = *(const uint4*)&Whh[(size_t)(n0 + rr) * I_ + k];
  }
#endif
  __syncthreads();

  const int m0   = wm * 16;
  const int col  = lane & 15;
  const int rowb = (lane >> 4) * 8;
  const int nn   = n0 + wn * 16 + col;
  unsigned short* Hs = (unsigned short*)&Lds[HOFF];

  for (int t = 0; t < S_; ++t) {
    const int s = dir ? (S_ - 1 - t) : t;
    const unsigned short* hprev = (t & 1) ? h1 : h0;
    unsigned short*       hnext = (t & 1) ? h0 : h1;

    // ---- stage h(t) into LDS (async global->LDS, ASYNCcnt-tracked) ----
    for (int idx = threadIdx.x; idx < 32 * (H_ / 8); idx += blockDim.x) {
      const int rr = idx >> 7;
      const int k  = (idx & 127) * 8;
      const unsigned short* src = hprev + (size_t)rr * H_ + k;
      unsigned short*       dst = Hs + rr * LDWP + padk(k);
#if HAVE_ASYNC_LDS
      __builtin_amdgcn_global_load_async_to_lds_b128(
          (g_v4i*)(const void*)src, (l_v4i*)(void*)dst, 0, 0);
#else
      *(uint4*)dst = *(const uint4*)src;
#endif
    }
#if HAVE_ASYNC_LDS
    asm volatile("s_wait_asynccnt 0" ::: "memory");
#endif
    __syncthreads();

    // ---- 16x16 tile GEMM: A (h) and B (W_hh) both from LDS ----
    v8f acc = {};
#pragma unroll 4
    for (int k0 = 0; k0 < H_; k0 += 32) {
      v16bf a = load_frag_lds(Hs,  m0,      k0, lane);
      v16bf b = load_frag_lds(Lds, wn * 16, k0, lane);
      acc = __builtin_amdgcn_wmma_f32_16x16x32_bf16(false, a, false, b, (short)0, acc, false, false);
    }

#pragma unroll
    for (int r = 0; r < 8; ++r) {
      const int bb = m0 + rowb + r;
      const float xpv = xp[((size_t)bb * S_ + s) * H_ + nn];
      const float hv  = tanhf(acc[r] + xpv);
      out[((size_t)bb * S_ + s) * (2 * H_) + (size_t)dir * H_ + nn] = hv;
      hnext[(size_t)bb * H_ + nn] = f2bf_raw(hv);
    }

    // ---- device-wide barrier over the 32 blocks of this direction ----
    __syncthreads();
    if (threadIdx.x == 0) {
      __threadfence();
      __hip_atomic_fetch_add(bar, 1u, __ATOMIC_RELEASE, __HIP_MEMORY_SCOPE_AGENT);
      const unsigned target = (unsigned)(t + 1) * 32u;
      while (__hip_atomic_load(bar, __ATOMIC_ACQUIRE, __HIP_MEMORY_SCOPE_AGENT) < target) {
        __builtin_amdgcn_s_sleep(1);
      }
    }
    __syncthreads();
  }
}

// ----------------------------------------------------------------- launcher
extern "C" void kernel_launch(void* const* d_in, const int* in_sizes, int n_in,
                              void* d_out, int out_size, void* d_ws, size_t ws_size,
                              hipStream_t stream)
{
  const float* x     = (const float*)d_in[0];
  const float* Wih_f = (const float*)d_in[1];
  const float* Whh_f = (const float*)d_in[2];
  const float* bih_f = (const float*)d_in[3];
  const float* bhh_f = (const float*)d_in[4];
  const float* Wih_b = (const float*)d_in[5];
  const float* Whh_b = (const float*)d_in[6];
  const float* bih_b = (const float*)d_in[7];
  const float* bhh_b = (const float*)d_in[8];
  float* out = (float*)d_out;

  char* ws = (char*)d_ws;
  size_t off = 0;
  auto carve = [&](size_t bytes) -> char* {
    char* p = ws + off;
    off = (off + bytes + 255) & ~(size_t)255;
    return p;
  };
  float*          xp_f    = (float*)carve((size_t)B_ * S_ * H_ * 4);
  float*          xp_b    = (float*)carve((size_t)B_ * S_ * H_ * 4);
  unsigned short* xbf     = (unsigned short*)carve((size_t)B_ * S_ * I_ * 2);
  unsigned short* wihf_bf = (unsigned short*)carve((size_t)H_ * I_ * 2);
  unsigned short* wihb_bf = (unsigned short*)carve((size_t)H_ * I_ * 2);
  unsigned short* whhf_bf = (unsigned short*)carve((size_t)H_ * I_ * 2);
  unsigned short* whhb_bf = (unsigned short*)carve((size_t)H_ * I_ * 2);
  unsigned short* hbuf    = (unsigned short*)carve((size_t)4 * B_ * H_ * 2);
  unsigned*       bars    = (unsigned*)carve(2 * sizeof(unsigned));
  (void)ws_size; (void)in_sizes; (void)n_in; (void)out_size;

  const int n_x = B_ * S_ * I_;
  const int n_w = H_ * I_;
  k_f32_to_bf16<<<(n_x + 255) / 256, 256, 0, stream>>>(x, xbf, n_x);
  k_f32_to_bf16<<<(n_w + 255) / 256, 256, 0, stream>>>(Wih_f, wihf_bf, n_w);
  k_f32_to_bf16<<<(n_w + 255) / 256, 256, 0, stream>>>(Wih_b, wihb_bf, n_w);
  k_f32_to_bf16<<<(n_w + 255) / 256, 256, 0, stream>>>(Whh_f, whhf_bf, n_w);
  k_f32_to_bf16<<<(n_w + 255) / 256, 256, 0, stream>>>(Whh_b, whhb_bf, n_w);

  (void)hipMemsetAsync(hbuf, 0, (size_t)B_ * H_ * 2, stream);                // dir0 h(0)
  (void)hipMemsetAsync(hbuf + 2 * B_ * H_, 0, (size_t)B_ * H_ * 2, stream);  // dir1 h(0)
  (void)hipMemsetAsync(bars, 0, 2 * sizeof(unsigned), stream);

  dim3 gA(128, 16, 2);
  k_gemm_xp<<<gA, 256, 0, stream>>>(xbf, wihf_bf, wihb_bf,
                                    bih_f, bhh_f, bih_b, bhh_b, xp_f, xp_b);

  const size_t lds_bytes = (size_t)2 * 32 * LDWP * sizeof(unsigned short); // ~130 KB / 320 KB WGP
  k_scan<<<64, 128, lds_bytes, stream>>>(whhf_bf, whhb_bf, xp_f, xp_b,
                                         hbuf, bars, out);
}
